// Conv_Block_one_43980465111270
// MI455X (gfx1250) — matrix operand order repaired and checked by reading, not compiled, not run on hardware
//
#include <hip/hip_runtime.h>
#include <hip/hip_bf16.h>

typedef __attribute__((ext_vector_type(16))) _Float16 v16h;
typedef __attribute__((ext_vector_type(8)))  _Float16 v8h;
typedef __attribute__((ext_vector_type(8)))  float    v8f;

union V16H { v16h v; v8h h[2]; };

// ---------------- geometry ----------------
static constexpr int IMG    = 512;
static constexpr int HW     = IMG * IMG;     // 262144 pixels
static constexpr int IC     = 64;
static constexpr int OCN    = 128;
static constexpr int TROWS  = 8;             // output rows per block
static constexpr int TCOLS  = 32;            // output cols per block
static constexpr int IN_R   = TROWS + 2;     // 10 staged input rows
static constexpr int IN_C   = TCOLS + 2;     // 34 staged input cols
static constexpr int CPAD   = 72;            // channel pitch in halves (144B, 16B-aligned, bank-rotated)

// ws layout: wT f16 [9][128][64] then scale[128], shift[128] f32
static constexpr int WT_ELEMS  = 9 * OCN * IC;          // 73728
static constexpr int WT_BYTES  = WT_ELEMS * 2;          // 147456

// ---------------- prep: weight transpose->f16 + BN fold ----------------
__global__ __launch_bounds__(256) void prep_kernel(
    const float* __restrict__ W, const float* __restrict__ gamma,
    const float* __restrict__ beta, const float* __restrict__ mean,
    const float* __restrict__ var, _Float16* __restrict__ wT,
    float* __restrict__ scale, float* __restrict__ shift)
{
    int idx = blockIdx.x * 256 + threadIdx.x;
    if (idx < WT_ELEMS) {
        int t   = idx / (OCN * IC);
        int rem = idx - t * (OCN * IC);
        int oc  = rem >> 6;
        int ic  = rem & 63;
        // W layout: [oc][ic][3][3] -> wT[t][oc][ic]
        wT[idx] = (_Float16)W[(oc * IC + ic) * 9 + t];
    }
    if (idx < OCN) {
        float inv = rsqrtf(var[idx] + 1e-5f);
        float s   = gamma[idx] * inv;
        scale[idx] = s;
        shift[idx] = beta[idx] - mean[idx] * s;
    }
}

__device__ __forceinline__ float postproc(float v, float sc, float sh, int mk) {
    float y = fmaf(v, sc, sh);
    y = (y >= 0.0f) ? y : 0.01f * y;
    return y * (float)mk;
}

// ---------------- main: implicit-GEMM conv + BN + LeakyReLU + mask ----------------
__global__ __launch_bounds__(256) void conv_main(
    const float* __restrict__ x, const int* __restrict__ mask,
    const _Float16* __restrict__ wT, const float* __restrict__ scale,
    const float* __restrict__ shift, float* __restrict__ out)
{
    __shared__ _Float16 lin[IN_R * IN_C * CPAD];   // 10*34*72 halves = 48960 B

    const int tid = threadIdx.x;
    const int bc  = blockIdx.x & 15;               // 16 col-blocks
    const int br  = blockIdx.x >> 4;               // 64 row-blocks
    const int r0  = br * TROWS;
    const int c0  = bc * TCOLS;

    // ---- stage input patch (reflect-padded) to LDS as f16 ----
    // 10*34*64 = 21760 elems = 85 full passes of 256 threads (no divergence)
    for (int idx = tid; idx < IC * IN_R * IN_C; idx += 256) {
        int ch  = idx / (IN_R * IN_C);
        int rem = idx - ch * (IN_R * IN_C);
        int row = rem / IN_C;
        int col = rem - row * IN_C;
        int gr  = r0 - 1 + row; gr = gr < 0 ? -gr : (gr > IMG - 1 ? 2 * (IMG - 1) - gr : gr);
        int gc  = c0 - 1 + col; gc = gc < 0 ? -gc : (gc > IMG - 1 ? 2 * (IMG - 1) - gc : gc);
        lin[(row * IN_C + col) * CPAD + ch] = (_Float16)x[ch * HW + gr * IMG + gc];
    }

    const int wv    = tid >> 5;          // wave id == oc-group
    const int ln    = tid & 31;
    const int lhalf = ln >> 4;           // 0: lanes 0-15, 1: lanes 16-31
    const int lmod  = ln & 15;
    const int oc    = wv * 16 + lmod;    // this lane's output channel (B/C "N")

    // ---- register-cache this wave's weight slice: 9 taps x 2 k-steps ----
    // B 32x16 f16 layout: lane N=L&15; lanes 0-15 hold K {0..7, 16..23}, lanes 16-31 hold K {8..15, 24..31}
    V16H b[9][2];
#pragma unroll
    for (int t = 0; t < 9; ++t) {
#pragma unroll
        for (int s = 0; s < 2; ++s) {
            const _Float16* p = wT + ((t * OCN + oc) * IC) + s * 32 + lhalf * 8;
            b[t][s].h[0] = *(const v8h*)(p);
            b[t][s].h[1] = *(const v8h*)(p + 16);
        }
    }

    const float sc = scale[oc];
    const float sh = shift[oc];

    __syncthreads();

    // ---- 16 pixel-groups of 16 pixels each (8 rows x 2 col-halves) ----
#pragma unroll 1
    for (int pg = 0; pg < 16; ++pg) {
        const int R  = pg >> 1;
        const int cb = (pg & 1) * 16;

        v8f acc = {0.f, 0.f, 0.f, 0.f, 0.f, 0.f, 0.f, 0.f};

#pragma unroll
        for (int t = 0; t < 9; ++t) {
            const int dy = t / 3 - 1;
            const int dx = t % 3 - 1;
            // A 16x32 f16 layout: lane M=lmod; halves map to
            //   lanes 0-15 : K {0..7, 16..23} (+k0) ; lanes 16-31 : K {8..15, 24..31} (+k0)
            const _Float16* ap =
                &lin[((1 + R + dy) * IN_C + (1 + cb + lmod + dx)) * CPAD + lhalf * 8];
            V16H a0, a1;
            a0.h[0] = *(const v8h*)(ap);        // K k0+{0..7}(+8*lhalf)
            a0.h[1] = *(const v8h*)(ap + 16);   // K k0+16+{0..7}(+8*lhalf)
            a1.h[0] = *(const v8h*)(ap + 32);   // k-step 1
            a1.h[1] = *(const v8h*)(ap + 48);
            acc = __builtin_amdgcn_wmma_f32_16x16x32_f16(
                false, a0.v, false, b[t][0].v, (short)0, acc, false, false);
            acc = __builtin_amdgcn_wmma_f32_16x16x32_f16(
                false, a1.v, false, b[t][1].v, (short)0, acc, false, false);
        }

        // ---- epilogue: C 16x16 f32 layout -> lane owns (oc, 8 consecutive pixels) ----
        const int grow = r0 + R;
        const int gcol = c0 + cb + lhalf * 8;
        const size_t base = (size_t)oc * HW + (size_t)grow * IMG + gcol;

        const int4 mk0 = *(const int4*)(mask + base);
        const int4 mk1 = *(const int4*)(mask + base + 4);

        float4 o0, o1;
        o0.x = postproc(acc[0], sc, sh, mk0.x);
        o0.y = postproc(acc[1], sc, sh, mk0.y);
        o0.z = postproc(acc[2], sc, sh, mk0.z);
        o0.w = postproc(acc[3], sc, sh, mk0.w);
        o1.x = postproc(acc[4], sc, sh, mk1.x);
        o1.y = postproc(acc[5], sc, sh, mk1.y);
        o1.z = postproc(acc[6], sc, sh, mk1.z);
        o1.w = postproc(acc[7], sc, sh, mk1.w);

        *(float4*)(out + base)     = o0;
        *(float4*)(out + base + 4) = o1;
    }
}

extern "C" void kernel_launch(void* const* d_in, const int* in_sizes, int n_in,
                              void* d_out, int out_size, void* d_ws, size_t ws_size,
                              hipStream_t stream) {
    const float* x     = (const float*)d_in[0];
    const float* W     = (const float*)d_in[1];
    const float* gamma = (const float*)d_in[2];
    const float* beta  = (const float*)d_in[3];
    const float* mean  = (const float*)d_in[4];
    const float* var   = (const float*)d_in[5];
    const int*   mask  = (const int*)d_in[6];
    float*       out   = (float*)d_out;

    _Float16* wT    = (_Float16*)d_ws;
    float*    scale = (float*)((char*)d_ws + WT_BYTES);
    float*    shift = scale + OCN;

    prep_kernel<<<(WT_ELEMS + 255) / 256, 256, 0, stream>>>(
        W, gamma, beta, mean, var, wT, scale, shift);

    // 64 row-blocks x 16 col-blocks = 1024 blocks, 256 threads (8 waves) each
    conv_main<<<64 * 16, 256, 0, stream>>>(x, mask, wT, scale, shift, out);
}
